// HetSANNConv_87514253623554
// MI455X (gfx1250) — matrix operand order repaired and checked by reading, not compiled, not run on hardware
//
#include <hip/hip_runtime.h>
#include <math.h>

// ---------------- problem constants ----------------
#define NN   50000
#define EE   800000
#define INF_ 128        // input dim
#define HD   256        // H*D
#define TT   5          // edge types
#define NHEAD 8
#define DIM   32
#define WAVES 4         // waves per block (wave32)

typedef __attribute__((ext_vector_type(16))) __bf16 v16b;
typedef __attribute__((ext_vector_type(8)))  float  v8f;
typedef __attribute__((ext_vector_type(4)))  unsigned int u32x4;
typedef __attribute__((ext_vector_type(4)))  float  f32x4;

union Frag { v16b v; u32x4 q[2]; };

__device__ __forceinline__ unsigned short f2bf(float f) {
  unsigned int u = __float_as_uint(f);
  unsigned int r = u + 0x7FFFu + ((u >> 16) & 1u);   // RNE
  return (unsigned short)(r >> 16);
}
__device__ __forceinline__ float bf2f(unsigned short h) {
  return __uint_as_float(((unsigned int)h) << 16);
}

// ---------------- prep: pack W[t][128][256] into WMMA-B lane layout ----------------
// layout: [t][kk(4)][n(16)][lane(32)][16 bf16]; lane: N=n*16+(l&15), Kbase=kk*32+(l>>4)*16
__global__ void pack_W_kernel(const float* __restrict__ W, unsigned short* __restrict__ Wp) {
  int idx = blockIdx.x * blockDim.x + threadIdx.x;
  if (idx >= TT * 4 * 16 * 32 * 16) return;
  int j  = idx & 1;
  int v  = (idx >> 1) & 7;
  int l  = (idx >> 4) & 31;
  int n  = (idx >> 9) & 15;
  int kk = (idx >> 13) & 3;
  int t  = idx >> 15;
  int K   = kk * 32 + (l >> 4) * 16 + 2 * v + j;
  int col = n * 16 + (l & 15);
  Wp[idx] = f2bf(W[(t * INF_ + K) * HD + col]);
}

// res_w is [256][128] (out,in); B needed as [in][out] -> B[K][col] = res_w[col][K]
__global__ void pack_R_kernel(const float* __restrict__ Rw, unsigned short* __restrict__ Rp) {
  int idx = blockIdx.x * blockDim.x + threadIdx.x;
  if (idx >= 4 * 16 * 32 * 16) return;
  int j  = idx & 1;
  int v  = (idx >> 1) & 7;
  int l  = (idx >> 4) & 31;
  int n  = (idx >> 9) & 15;
  int kk = (idx >> 13) & 3;
  int K   = kk * 32 + (l >> 4) * 16 + 2 * v + j;
  int col = n * 16 + (l & 15);
  Rp[idx] = f2bf(Rw[col * INF_ + K]);
}

// alr[t][k][head] = sum_d (a_l[t][k][head*32+d] + a_r[t][k][head*32+d])
__global__ void reduce_alr_kernel(const float* __restrict__ al,
                                  const float* __restrict__ ar,
                                  float* __restrict__ alr) {
  int idx = blockIdx.x * blockDim.x + threadIdx.x;
  if (idx >= TT * HD * NHEAD) return;
  int head = idx & 7;
  int k    = (idx >> 3) & 255;
  int t    = idx >> 11;
  const float* pl = al + ((size_t)(t * HD + k)) * HD + head * DIM;
  const float* pr = ar + ((size_t)(t * HD + k)) * HD + head * DIM;
  float s = 0.f;
  #pragma unroll
  for (int d = 0; d < DIM; ++d) s += pl[d] + pr[d];
  alr[idx] = s;
}

// ---------------- fused edge kernel ----------------
// wave handles 16 edges: gather feat -> masked typed WMMA GEMM (h = feat@W[t]) ->
// attention (reduced alr) -> atomic scatter alpha into emb[N][256]
__global__ void __launch_bounds__(WAVES * 32)
hetsann_edge_kernel(const float* __restrict__ feat,
                    const float* __restrict__ eweight,
                    const int*   __restrict__ src,
                    const int*   __restrict__ dst,
                    const int*   __restrict__ ety,
                    const unsigned short* __restrict__ Wp,
                    const float* __restrict__ alr,
                    float* __restrict__ emb) {
  __shared__ __align__(16) unsigned short sfeat[WAVES][16][INF_];
  __shared__ __align__(16) unsigned short shh[WAVES][16][HD];
  __shared__ float satt[WAVES][16][NHEAD];

  const int tid  = threadIdx.x;
  const int w    = tid >> 5;
  const int l    = tid & 31;
  const int row  = l & 15;
  const int half = l >> 4;
  const int ebase = (blockIdx.x * WAVES + w) * 16;   // E = 12500*4*16 exactly

  // ---- stage 16 gathered feature rows to LDS as bf16 (each lane: 4 cols/row) ----
  for (int r = 0; r < 16; ++r) {
    int s = src[ebase + r];
    f32x4 f4 = *reinterpret_cast<const f32x4*>(feat + (size_t)s * INF_ + l * 4);
    unsigned int lo = (unsigned int)f2bf(f4.x) | ((unsigned int)f2bf(f4.y) << 16);
    unsigned int hi = (unsigned int)f2bf(f4.z) | ((unsigned int)f2bf(f4.w) << 16);
    unsigned int* p = reinterpret_cast<unsigned int*>(&sfeat[w][r][l * 4]);
    p[0] = lo; p[1] = hi;
  }
  const int myt = ety[ebase + row];

  v8f acc[16];
  #pragma unroll
  for (int n = 0; n < 16; ++n) {
    #pragma unroll
    for (int i = 0; i < 8; ++i) acc[n][i] = 0.f;
  }

  const u32x4 z4 = {0u, 0u, 0u, 0u};
  for (int kk = 0; kk < 4; ++kk) {
    Frag af;
    const unsigned short* ap = &sfeat[w][row][kk * 32 + half * 16];
    af.q[0] = *reinterpret_cast<const u32x4*>(ap);
    af.q[1] = *reinterpret_cast<const u32x4*>(ap + 8);
    for (int t = 0; t < TT; ++t) {
      const bool on = (myt == t);
      Frag at;
      at.q[0] = on ? af.q[0] : z4;
      at.q[1] = on ? af.q[1] : z4;
      const unsigned short* wb = Wp + (size_t)(((t * 4 + kk) * 16) * 32 + l) * 16;
      #pragma unroll
      for (int n = 0; n < 16; ++n) {
        Frag bf_;
        const unsigned short* bp = wb + n * 512;
        bf_.q[0] = *reinterpret_cast<const u32x4*>(bp);
        bf_.q[1] = *reinterpret_cast<const u32x4*>(bp + 8);
        acc[n] = __builtin_amdgcn_wmma_f32_16x16x32_bf16(
            false, at.v, false, bf_.v, (short)0, acc[n], false, false);
      }
    }
  }

  // ---- spill h tile (16 edges x 256) to LDS as bf16 ----
  #pragma unroll
  for (int n = 0; n < 16; ++n) {
    #pragma unroll
    for (int j = 0; j < 8; ++j) {
      shh[w][half * 8 + j][n * 16 + row] = f2bf(acc[n][j]);
    }
  }

  // ---- attention: s[e][head] = sum_k h[e][k]*alr[t_e][k][head]; 4 (e,head) pairs/lane ----
  #pragma unroll
  for (int i = 0; i < 4; ++i) {
    int p = i * 32 + l;
    int e = p >> 3, head = p & 7;
    int te = ety[ebase + e];
    const float* c = alr + te * (HD * NHEAD) + head;
    float s = 0.f;
    #pragma unroll 8
    for (int k = 0; k < HD; ++k) s += bf2f(shh[w][e][k]) * c[k * NHEAD];
    float lr = s > 0.f ? s : 0.2f * s;
    float sg = 1.f / (1.f + __expf(-lr));
    satt[w][e][head] = sg * eweight[ebase + e];
  }

  // ---- scatter: emb[dst][ (d)*8 + head ] += h[e][head*32+d] * att[e][head] ----
  for (int r = 0; r < 16; ++r) {
    int node = dst[ebase + r];
    float* ob = emb + (size_t)node * HD;
    #pragma unroll
    for (int m = 0; m < 8; ++m) {      // head = m, col k = m*32 + l, out pos = l*8 + m
      float v = bf2f(shh[w][r][m * 32 + l]) * satt[w][r][m];
      atomicAdd(ob + l * 8 + m, v);
    }
  }
}

// ---------------- node finalize: out = elu(emb + feat @ res_w^T + b) via WMMA ----------------
__global__ void __launch_bounds__(WAVES * 32)
hetsann_node_kernel(const float* __restrict__ feat,
                    const float* __restrict__ emb,
                    const unsigned short* __restrict__ Rp,
                    const float* __restrict__ res_b,
                    float* __restrict__ out) {
  __shared__ __align__(16) unsigned short sfeat[WAVES][16][INF_];
  const int tid  = threadIdx.x;
  const int w    = tid >> 5;
  const int l    = tid & 31;
  const int row  = l & 15;
  const int half = l >> 4;
  const int tile = blockIdx.x * WAVES + w;
  if (tile >= NN / 16) return;          // uniform per wave; no barriers used
  const int nbase = tile * 16;

  for (int r = 0; r < 16; ++r) {
    f32x4 f4 = *reinterpret_cast<const f32x4*>(feat + (size_t)(nbase + r) * INF_ + l * 4);
    unsigned int lo = (unsigned int)f2bf(f4.x) | ((unsigned int)f2bf(f4.y) << 16);
    unsigned int hi = (unsigned int)f2bf(f4.z) | ((unsigned int)f2bf(f4.w) << 16);
    unsigned int* p = reinterpret_cast<unsigned int*>(&sfeat[w][r][l * 4]);
    p[0] = lo; p[1] = hi;
  }

  v8f acc[16];
  #pragma unroll
  for (int n = 0; n < 16; ++n) {
    #pragma unroll
    for (int i = 0; i < 8; ++i) acc[n][i] = 0.f;
  }

  for (int kk = 0; kk < 4; ++kk) {
    Frag af;
    const unsigned short* ap = &sfeat[w][row][kk * 32 + half * 16];
    af.q[0] = *reinterpret_cast<const u32x4*>(ap);
    af.q[1] = *reinterpret_cast<const u32x4*>(ap + 8);
    const unsigned short* rb = Rp + (size_t)((kk * 16) * 32 + l) * 16;
    #pragma unroll
    for (int n = 0; n < 16; ++n) {
      Frag bf_;
      const unsigned short* bp = rb + n * 512;
      bf_.q[0] = *reinterpret_cast<const u32x4*>(bp);
      bf_.q[1] = *reinterpret_cast<const u32x4*>(bp + 8);
      acc[n] = __builtin_amdgcn_wmma_f32_16x16x32_bf16(
          false, af.v, false, bf_.v, (short)0, acc[n], false, false);
    }
  }

  #pragma unroll
  for (int n = 0; n < 16; ++n) {
    #pragma unroll
    for (int j = 0; j < 8; ++j) {
      int node = nbase + half * 8 + j;
      int col  = n * 16 + row;
      float v = acc[n][j] + emb[(size_t)node * HD + col] + res_b[col];
      out[(size_t)node * HD + col] = v > 0.f ? v : (__expf(v) - 1.f);
    }
  }
}

// ---------------- launch ----------------
extern "C" void kernel_launch(void* const* d_in, const int* in_sizes, int n_in,
                              void* d_out, int out_size, void* d_ws, size_t ws_size,
                              hipStream_t stream) {
  const float* feat    = (const float*)d_in[0];
  const float* eweight = (const float*)d_in[1];
  const int*   src     = (const int*)d_in[2];
  const int*   dst     = (const int*)d_in[3];
  const int*   ety     = (const int*)d_in[4];
  // d_in[5] = ntype_idxs (unused by reference)
  const float* W       = (const float*)d_in[6];
  const float* a_l     = (const float*)d_in[7];
  const float* a_r     = (const float*)d_in[8];
  const float* res_w   = (const float*)d_in[9];
  const float* res_b   = (const float*)d_in[10];
  float* out = (float*)d_out;

  char* ws = (char*)d_ws;
  unsigned short* Wp  = (unsigned short*)(ws);            // 327,680 B
  unsigned short* Rp  = (unsigned short*)(ws + 327680);   //  65,536 B
  float*          alr = (float*)(ws + 393216);            //  40,960 B
  float*          emb = (float*)(ws + 434176);            // 51.2 MB

  hipMemsetAsync(emb, 0, (size_t)NN * HD * sizeof(float), stream);
  pack_W_kernel<<<(TT * 4 * 16 * 32 * 16 + 255) / 256, 256, 0, stream>>>(W, Wp);
  pack_R_kernel<<<(4 * 16 * 32 * 16 + 255) / 256, 256, 0, stream>>>(res_w, Rp);
  reduce_alr_kernel<<<(TT * HD * NHEAD + 255) / 256, 256, 0, stream>>>(a_l, a_r, alr);

  hetsann_edge_kernel<<<EE / (WAVES * 16), WAVES * 32, 0, stream>>>(
      feat, eweight, src, dst, ety, Wp, alr, emb);
  hetsann_node_kernel<<<(NN / 16 + WAVES - 1) / WAVES, WAVES * 32, 0, stream>>>(
      feat, emb, Rp, res_b, out);
}
